// MoE_88416196755615
// MI455X (gfx1250) — compile-verified, matrix operand order checked by reading, not run on hardware
//
#include <hip/hip_runtime.h>
#include <hip/hip_bf16.h>
#include <cstdint>
#include <cstddef>

// ---------------- types for WMMA fragments ----------------
typedef __bf16 bf16;
typedef __attribute__((ext_vector_type(16))) __bf16 v16bf;
typedef __attribute__((ext_vector_type(8)))  float  v8f;
typedef __attribute__((ext_vector_type(4)))  unsigned int v4u;
typedef int v4i_vs __attribute__((vector_size(16)));   // matches builtin param type

union FragBF {            // 16 bf16 = 32 bytes = 8 VGPRs
    v16bf v;
    v4u   u[2];
};

#define AS1 __attribute__((address_space(1)))
#define AS3 __attribute__((address_space(3)))

#if __has_builtin(__builtin_amdgcn_global_load_async_to_lds_b128)
#define HAVE_ASYNC_LDS 1
#else
#define HAVE_ASYNC_LDS 0
#endif

static __device__ __forceinline__ void async_wait_all() {
#if HAVE_ASYNC_LDS
#if __has_builtin(__builtin_amdgcn_s_wait_asynccnt)
    __builtin_amdgcn_s_wait_asynccnt(0);
#else
    asm volatile("s_wait_asynccnt 0" ::: "memory");
#endif
#endif
}

static __device__ __forceinline__ void stage16(const bf16* g, bf16* l) {
#if HAVE_ASYNC_LDS
    __builtin_amdgcn_global_load_async_to_lds_b128(
        (AS1 v4i_vs*)(g), (AS3 v4i_vs*)(l), 0, 0);
#else
    *(v4u*)l = *(const v4u*)g;
#endif
}

// ---------------- problem constants ----------------
#define TOKS   16384      // B*S
#define DIM    768        // D
#define NEXP   8          // E
#define FDIM   469        // F
#define FPAD   480        // F padded to multiple of 32 (K for GEMM2)
#define NT32   (TOKS / 32)

#define AROW1  (DIM + 8)   // 776: LDS row stride for GEMM1 A tile (bank-rotated)
#define AROW2  (FPAD + 8)  // 488: LDS row stride for GEMM2 A tile

// ---------------- workspace layout (bytes) ----------------
#define WS_XB    ((size_t)0)                      // bf16 [TOKS][DIM]      25,165,824
#define WS_W1B   ((size_t)25165824)               // bf16 [E][FPAD][DIM]    5,898,240
#define WS_W2B   ((size_t)31064064)               // bf16 [E][FPAD][DIM]    5,898,240
#define WS_W3B   ((size_t)36962304)               // bf16 [E][DIM][FPAD]    5,898,240
#define WS_HB    ((size_t)42860544)               // bf16 [TOKS][FPAD]     15,728,640
#define WS_CNT   ((size_t)58589184)               // int  [E]
#define WS_TOK   ((size_t)58589440)               // int  [E][TOKS]           524,288

static __device__ __forceinline__ float silu_f(float v) {
    return v / (1.0f + __expf(-v));
}

// ---------------- prep kernels ----------------
__global__ __launch_bounds__(32) void zero_counts_kernel(int* __restrict__ counts) {
    if (threadIdx.x < NEXP) counts[threadIdx.x] = 0;
}

__global__ __launch_bounds__(256) void cvt_x_kernel(const float* __restrict__ src,
                                                    bf16* __restrict__ dst) {
    size_t i = (size_t)blockIdx.x * 256 + threadIdx.x;   // < TOKS*DIM
    dst[i] = (bf16)src[i];
}

// W1/W2: fp32 [E][F][D] -> bf16 [E][FPAD][D] with zero pad rows
__global__ __launch_bounds__(256) void cvt_w12_kernel(const float* __restrict__ src,
                                                      bf16* __restrict__ dst) {
    size_t i = (size_t)blockIdx.x * 256 + threadIdx.x;   // < E*FPAD*DIM
    int d  = (int)(i % DIM);
    int ef = (int)(i / DIM);
    int f  = ef % FPAD;
    int e  = ef / FPAD;
    float v = (f < FDIM) ? src[((size_t)e * FDIM + f) * DIM + d] : 0.0f;
    dst[i] = (bf16)v;
}

// W3: fp32 [E][D][F] -> bf16 [E][D][FPAD] with zero pad cols
__global__ __launch_bounds__(256) void cvt_w3_kernel(const float* __restrict__ src,
                                                     bf16* __restrict__ dst) {
    size_t i = (size_t)blockIdx.x * 256 + threadIdx.x;   // < E*DIM*FPAD
    int f  = (int)(i % FPAD);
    size_t ed = i / FPAD;                                // e*DIM + d
    float v = (f < FDIM) ? src[ed * FDIM + f] : 0.0f;
    dst[i] = (bf16)v;
}

// ---------------- gating: one wave32 per token ----------------
__global__ __launch_bounds__(256) void gate_kernel(const float* __restrict__ x,
                                                   const float* __restrict__ Wg,
                                                   const float* __restrict__ bg,
                                                   int* __restrict__ counts,
                                                   int* __restrict__ tokIds) {
    __shared__ float sWg[NEXP * DIM];                    // 24 KB of the 320 KB WGP LDS
    for (int i = threadIdx.x; i < NEXP * DIM; i += 256) sWg[i] = Wg[i];
    __syncthreads();

    const int wave = threadIdx.x >> 5;
    const int lane = threadIdx.x & 31;
    const int t = blockIdx.x * 8 + wave;                 // token id

    const float* xr = x + (size_t)t * DIM;
    float acc[NEXP];
#pragma unroll
    for (int e = 0; e < NEXP; ++e) acc[e] = 0.0f;

    for (int d = lane; d < DIM; d += 32) {
        float xv = xr[d];
#pragma unroll
        for (int e = 0; e < NEXP; ++e) acc[e] += xv * sWg[e * DIM + d];
    }
    // wave32 reduction
#pragma unroll
    for (int off = 16; off > 0; off >>= 1) {
#pragma unroll
        for (int e = 0; e < NEXP; ++e) acc[e] += __shfl_xor(acc[e], off, 32);
    }
    if (lane == 0) {
        float best = acc[0] + bg[0];
        int bi = 0;
#pragma unroll
        for (int e = 1; e < NEXP; ++e) {
            float v = acc[e] + bg[e];
            if (v > best) { best = v; bi = e; }
        }
        int pos = atomicAdd(&counts[bi], 1);
        tokIds[bi * TOKS + pos] = t;                     // top-1 prob == 1.0
    }
}

// ---------------- GEMM1 + SwiGLU: h = silu(x@W1^T) * (x@W2^T) ----------------
// block = (expert, 32-token tile); 10 waves, each owns 3 N-tiles (10*48 = FPAD).
// A tile (32 gathered token rows) staged to LDS via async-to-LDS; B streams from L2.
__global__ __launch_bounds__(320) void gemm1_kernel(const bf16* __restrict__ xb,
                                                    const bf16* __restrict__ w1b,
                                                    const bf16* __restrict__ w2b,
                                                    const int*  __restrict__ counts,
                                                    const int*  __restrict__ tokIds,
                                                    bf16* __restrict__ hb) {
    const int e    = blockIdx.x / NT32;
    const int tile = blockIdx.x % NT32;
    const int cnt  = counts[e];
    if (tile * 32 >= cnt) return;

    __shared__ __align__(16) bf16 sA[32 * AROW1];        // 48.5 KB

    // ---- stage 32 gathered A rows (coalesced 16B chunks, async to LDS) ----
    const int CHR = DIM / 8;                             // 96 16B-chunks per row
    for (int c = threadIdx.x; c < 32 * CHR; c += 320) {
        int row  = c / CHR;
        int col8 = c % CHR;
        int ri = tile * 32 + row;
        int tk = tokIds[e * TOKS + (ri < cnt ? ri : cnt - 1)];
        stage16(xb + (size_t)tk * DIM + col8 * 8, sA + row * AROW1 + col8 * 8);
    }
    async_wait_all();
    __syncthreads();

    const int wave = threadIdx.x >> 5;
    const int lane = threadIdx.x & 31;
    const int m16  = lane & 15;            // fragment row/col index
    const int g    = lane >> 4;            // K-halving group (ISA 16-bit layout)
    const int nb   = wave * 48;            // this wave's N base

    v8f c1[2][3], c2[2][3];
#pragma unroll
    for (int mt = 0; mt < 2; ++mt)
#pragma unroll
        for (int j = 0; j < 3; ++j) { c1[mt][j] = (v8f)0.0f; c2[mt][j] = (v8f)0.0f; }

    for (int k = 0; k < DIM; k += 32) {
        FragBF a[2];
#pragma unroll
        for (int mt = 0; mt < 2; ++mt) {
            const bf16* ar = sA + (mt * 16 + m16) * AROW1 + k + 8 * g;
            a[mt].u[0] = *(const v4u*)(ar);
            a[mt].u[1] = *(const v4u*)(ar + 16);
        }
#pragma unroll
        for (int j = 0; j < 3; ++j) {
            const int n = nb + j * 16 + m16;             // B column = W row
            const bf16* b1r = w1b + ((size_t)e * FPAD + n) * DIM + k + 8 * g;
            const bf16* b2r = w2b + ((size_t)e * FPAD + n) * DIM + k + 8 * g;
            FragBF b1, b2;
            b1.u[0] = *(const v4u*)(b1r);
            b1.u[1] = *(const v4u*)(b1r + 16);
            b2.u[0] = *(const v4u*)(b2r);
            b2.u[1] = *(const v4u*)(b2r + 16);
#pragma unroll
            for (int mt = 0; mt < 2; ++mt) {
                c1[mt][j] = __builtin_amdgcn_wmma_f32_16x16x32_bf16(
                                false, a[mt].v, false, b1.v, (short)0, c1[mt][j], false, false);
                c2[mt][j] = __builtin_amdgcn_wmma_f32_16x16x32_bf16(
                                false, a[mt].v, false, b2.v, (short)0, c2[mt][j], false, false);
            }
        }
    }
    // D layout: VGPR r holds M = r + 8*g, N = lane&15
#pragma unroll
    for (int mt = 0; mt < 2; ++mt) {
#pragma unroll
        for (int r = 0; r < 8; ++r) {
            int m = mt * 16 + r + 8 * g;
            int ri = tile * 32 + m;
            if (ri < cnt) {
                int tk = tokIds[e * TOKS + ri];
                bf16* hrow = hb + (size_t)tk * FPAD;
#pragma unroll
                for (int j = 0; j < 3; ++j) {
                    float hv = silu_f(c1[mt][j][r]) * c2[mt][j][r];
                    hrow[nb + j * 16 + m16] = (bf16)hv;
                }
            }
        }
    }
}

// ---------------- GEMM2: out = h @ W3^T ----------------
// block = (expert, 32-token tile); 8 waves, each owns 6 N-tiles (8*96 = DIM)
__global__ __launch_bounds__(256) void gemm2_kernel(const bf16* __restrict__ hb,
                                                    const bf16* __restrict__ w3b,
                                                    const int*  __restrict__ counts,
                                                    const int*  __restrict__ tokIds,
                                                    float* __restrict__ out) {
    const int e    = blockIdx.x / NT32;
    const int tile = blockIdx.x % NT32;
    const int cnt  = counts[e];
    if (tile * 32 >= cnt) return;

    __shared__ __align__(16) bf16 sA[32 * AROW2];        // 30.5 KB

    const int CHR = FPAD / 8;                            // 60 16B-chunks per row
    for (int c = threadIdx.x; c < 32 * CHR; c += 256) {
        int row  = c / CHR;
        int col8 = c % CHR;
        int ri = tile * 32 + row;
        int tk = tokIds[e * TOKS + (ri < cnt ? ri : cnt - 1)];
        stage16(hb + (size_t)tk * FPAD + col8 * 8, sA + row * AROW2 + col8 * 8);
    }
    async_wait_all();
    __syncthreads();

    const int wave = threadIdx.x >> 5;
    const int lane = threadIdx.x & 31;
    const int m16  = lane & 15;
    const int g    = lane >> 4;
    const int nb   = wave * 96;

    v8f c[2][6];
#pragma unroll
    for (int mt = 0; mt < 2; ++mt)
#pragma unroll
        for (int j = 0; j < 6; ++j) c[mt][j] = (v8f)0.0f;

    for (int k = 0; k < FPAD; k += 32) {
        FragBF a[2];
#pragma unroll
        for (int mt = 0; mt < 2; ++mt) {
            const bf16* ar = sA + (mt * 16 + m16) * AROW2 + k + 8 * g;
            a[mt].u[0] = *(const v4u*)(ar);
            a[mt].u[1] = *(const v4u*)(ar + 16);
        }
#pragma unroll
        for (int j = 0; j < 6; ++j) {
            const int n = nb + j * 16 + m16;             // B column = W3 row (d)
            const bf16* br = w3b + ((size_t)e * DIM + n) * FPAD + k + 8 * g;
            FragBF b;
            b.u[0] = *(const v4u*)(br);
            b.u[1] = *(const v4u*)(br + 16);
#pragma unroll
            for (int mt = 0; mt < 2; ++mt) {
                c[mt][j] = __builtin_amdgcn_wmma_f32_16x16x32_bf16(
                               false, a[mt].v, false, b.v, (short)0, c[mt][j], false, false);
            }
        }
    }
#pragma unroll
    for (int mt = 0; mt < 2; ++mt) {
#pragma unroll
        for (int r = 0; r < 8; ++r) {
            int m = mt * 16 + r + 8 * g;
            int ri = tile * 32 + m;
            if (ri < cnt) {
                int tk = tokIds[e * TOKS + ri];
                float* orow = out + (size_t)tk * DIM;
#pragma unroll
                for (int j = 0; j < 6; ++j) {
                    orow[nb + j * 16 + m16] = c[mt][j][r];
                }
            }
        }
    }
}

// ---------------- launcher ----------------
extern "C" void kernel_launch(void* const* d_in, const int* in_sizes, int n_in,
                              void* d_out, int out_size, void* d_ws, size_t ws_size,
                              hipStream_t stream) {
    const float* x  = (const float*)d_in[0];   // [B,S,D]
    const float* Wg = (const float*)d_in[1];   // [E,D]
    const float* bg = (const float*)d_in[2];   // [E]
    const float* W1 = (const float*)d_in[3];   // [E,F,D]
    const float* W2 = (const float*)d_in[4];   // [E,F,D]
    const float* W3 = (const float*)d_in[5];   // [E,D,F]
    float* out = (float*)d_out;

    char* ws = (char*)d_ws;
    bf16* xb     = (bf16*)(ws + WS_XB);
    bf16* w1b    = (bf16*)(ws + WS_W1B);
    bf16* w2b    = (bf16*)(ws + WS_W2B);
    bf16* w3b    = (bf16*)(ws + WS_W3B);
    bf16* hb     = (bf16*)(ws + WS_HB);
    int*  counts = (int*)(ws + WS_CNT);
    int*  tokIds = (int*)(ws + WS_TOK);

    zero_counts_kernel<<<1, 32, 0, stream>>>(counts);

    cvt_x_kernel<<<(TOKS * DIM) / 256, 256, 0, stream>>>(x, xb);
    cvt_w12_kernel<<<(NEXP * FPAD * DIM) / 256, 256, 0, stream>>>(W1, w1b);
    cvt_w12_kernel<<<(NEXP * FPAD * DIM) / 256, 256, 0, stream>>>(W2, w2b);
    cvt_w3_kernel<<<(NEXP * DIM * FPAD) / 256, 256, 0, stream>>>(W3, w3b);

    gate_kernel<<<TOKS / 8, 256, 0, stream>>>(x, Wg, bg, counts, tokIds);

    gemm1_kernel<<<NEXP * NT32, 320, 0, stream>>>(xb, w1b, w2b, counts, tokIds, hb);
    gemm2_kernel<<<NEXP * NT32, 256, 0, stream>>>(hb, w3b, counts, tokIds, out);
}
